// _TriangularUpdate_15161234555332
// MI455X (gfx1250) — matrix operand order repaired and checked by reading, not compiled, not run on hardware
//
#include <hip/hip_runtime.h>

#define NDIM 512
#define DDIM 128
#define NNROWS (NDIM * NDIM)          // 262144
#define EPSV 1e-5f

typedef __bf16 bf16_t;
typedef __attribute__((ext_vector_type(16))) __bf16          v16bf;
typedef __attribute__((ext_vector_type(8)))  __bf16          v8bf;
typedef __attribute__((ext_vector_type(8)))  float           v8f;
typedef __attribute__((ext_vector_type(8)))  unsigned short  v8u16;
typedef __attribute__((ext_vector_type(4)))  unsigned short  v4u16;

// ---------------- scalar helpers ----------------

static __device__ __forceinline__ unsigned short f2bf(float f) {
    unsigned u = __builtin_bit_cast(unsigned, f);
    // round-to-nearest-even
    u += 0x7FFFu + ((u >> 16) & 1u);
    return (unsigned short)(u >> 16);
}

static __device__ __forceinline__ float bf2f(unsigned short u) {
    return __builtin_bit_cast(float, ((unsigned)u) << 16);
}

static __device__ __forceinline__ float sigf(float x) {
    return 1.0f / (1.0f + __expf(-x));
}

// ---------------- WMMA fragment helpers ----------------
// A-matrix 16x32 bf16: lane L holds row M=L&15;
//   lanes 0-15 hold K=0..15, lanes 16-31 hold K=16..31 (matches B's k order).
// -> per lane: one contiguous 16-element (32B) chunk at K = half*16.
static __device__ __forceinline__ v16bf load_a_frag(const bf16_t* base, int ld, int lane) {
    int half = lane >> 4;
    int r    = lane & 15;
    return *(const v16bf*)(base + (size_t)r * ld + half * 16);
}

// B-matrix 32x16 bf16: lane L holds column N=L&15;
//   lanes 0-15 hold K=0..15, lanes 16-31 hold K=16..31 (one contiguous 32B chunk).
static __device__ __forceinline__ v16bf load_b_frag(const bf16_t* base, int ld, int lane) {
    int half = lane >> 4;
    int c    = lane & 15;
    return *(const v16bf*)(base + (size_t)c * ld + half * 16);
}

static __device__ __forceinline__ v8f wmma_bf16(v16bf a, v16bf b, v8f c) {
    return __builtin_amdgcn_wmma_f32_16x16x32_bf16(
        /*neg_a=*/false, a, /*neg_b=*/false, b,
        /*c_mod=*/(short)0, c, /*reuse_a=*/false, /*reuse_b=*/false);
}

// acc over K=128 with preloaded A fragments; Wtile points at (col0, k=0), ld = 128
static __device__ __forceinline__ v8f gemm_row128(const v16bf a[4], const bf16_t* Wtile, int lane) {
    v8f acc = {0.f, 0.f, 0.f, 0.f, 0.f, 0.f, 0.f, 0.f};
#pragma unroll
    for (int ks = 0; ks < 4; ++ks) {
        v16bf b = load_b_frag(Wtile + ks * 32, DDIM, lane);
        acc = wmma_bf16(a[ks], b, acc);
    }
    return acc;
}

// ---------------- K0: fp32 -> bf16 weight conversion (6 x 128x128) ----------------
__global__ void k_cvt_weights(const float* w0, const float* w1, const float* w2,
                              const float* w3, const float* w4, const float* w5,
                              bf16_t* dst) {
    int idx = blockIdx.x * blockDim.x + threadIdx.x;    // 0 .. 6*16384-1
    int mat = idx >> 14;
    int off = idx & 16383;
    const float* s = (mat == 0) ? w0 : (mat == 1) ? w1 : (mat == 2) ? w2
                   : (mat == 3) ? w3 : (mat == 4) ? w4 : w5;
    ((unsigned short*)dst)[idx] = f2bf(s[off]);
}

// ---------------- K1/K4a: LayerNorm over D=128, fp32 in -> bf16 out ----------------
// one wave per row; lane loads float4 (512B/row coalesced); shfl_xor width-32 reduce
__global__ void k_layernorm_bf16(const float* __restrict__ x,
                                 const float* __restrict__ w,
                                 const float* __restrict__ b,
                                 bf16_t* __restrict__ y) {
    int wv   = threadIdx.x >> 5;
    int lane = threadIdx.x & 31;
    size_t row = (size_t)blockIdx.x * 8 + wv;
    const float4 xv = *(const float4*)(x + row * DDIM + lane * 4);
    float s  = xv.x + xv.y + xv.z + xv.w;
    float s2 = xv.x * xv.x + xv.y * xv.y + xv.z * xv.z + xv.w * xv.w;
#pragma unroll
    for (int m = 16; m >= 1; m >>= 1) {
        s  += __shfl_xor(s, m, 32);
        s2 += __shfl_xor(s2, m, 32);
    }
    float mean = s * (1.0f / DDIM);
    float var  = s2 * (1.0f / DDIM) - mean * mean;
    float inv  = rsqrtf(var + EPSV);
    float4 wv4 = *(const float4*)(w + lane * 4);
    float4 bv4 = *(const float4*)(b + lane * 4);
    v4u16 o;
    o[0] = f2bf((xv.x - mean) * inv * wv4.x + bv4.x);
    o[1] = f2bf((xv.y - mean) * inv * wv4.y + bv4.y);
    o[2] = f2bf((xv.z - mean) * inv * wv4.z + bv4.z);
    o[3] = f2bf((xv.w - mean) * inv * wv4.w + bv4.w);
    *(v4u16*)((unsigned short*)y + row * DDIM + lane * 4) = o;
}

// ---------------- K2: fused 5-way gated projections ----------------
// block = 8 waves; wave w owns d-slice [w*16, w*16+16) for all 5 weights.
// left/right stored TRANSPOSED: [d][row] bf16 (each lane: one 16B contiguous store).
// gate stored natural layout [row][d] bf16.
__global__ void k_proj_gated(const bf16_t* __restrict__ p,
                             const bf16_t* __restrict__ wbase,
                             const float* __restrict__ lpb, const float* __restrict__ lgb,
                             const float* __restrict__ rpb, const float* __restrict__ rgb,
                             const float* __restrict__ ogb,
                             const float* __restrict__ mask,
                             bf16_t* __restrict__ leftT, bf16_t* __restrict__ rightT,
                             bf16_t* __restrict__ gate) {
    int wv   = threadIdx.x >> 5;
    int lane = threadIdx.x & 31;
    int half = lane >> 4;
    int nn   = lane & 15;
    int d0   = wv * 16;
    size_t row0 = (size_t)blockIdx.x * 16;

    // preload shared A fragments (16 rows x 128 K)
    v16bf a[4];
#pragma unroll
    for (int ks = 0; ks < 4; ++ks)
        a[ks] = load_a_frag(p + row0 * DDIM + ks * 32, DDIM, lane);

    float mk[8];
#pragma unroll
    for (int r = 0; r < 8; ++r) mk[r] = mask[row0 + 8 * half + r];

    const bf16_t* Wlp = wbase + (size_t)0 * 16384;
    const bf16_t* Wlg = wbase + (size_t)1 * 16384;
    const bf16_t* Wrp = wbase + (size_t)2 * 16384;
    const bf16_t* Wrg = wbase + (size_t)3 * 16384;
    const bf16_t* Wog = wbase + (size_t)5 * 16384;

    // left = (p@lpw.T + lpb) * sigmoid(p@lgw.T + lgb) * mask
    {
        v8f aP = gemm_row128(a, Wlp + (size_t)d0 * DDIM, lane);
        v8f aG = gemm_row128(a, Wlg + (size_t)d0 * DDIM, lane);
        float bp = lpb[d0 + nn], bg = lgb[d0 + nn];
        v8u16 o;
#pragma unroll
        for (int r = 0; r < 8; ++r)
            o[r] = f2bf((aP[r] + bp) * sigf(aG[r] + bg) * mk[r]);
        *(v8u16*)((unsigned short*)leftT + (size_t)(d0 + nn) * NNROWS + row0 + 8 * half) = o;
    }
    // right = (p@rpw.T + rpb) * sigmoid(p@rgw.T + rgb) * mask
    {
        v8f aP = gemm_row128(a, Wrp + (size_t)d0 * DDIM, lane);
        v8f aG = gemm_row128(a, Wrg + (size_t)d0 * DDIM, lane);
        float bp = rpb[d0 + nn], bg = rgb[d0 + nn];
        v8u16 o;
#pragma unroll
        for (int r = 0; r < 8; ++r)
            o[r] = f2bf((aP[r] + bp) * sigf(aG[r] + bg) * mk[r]);
        *(v8u16*)((unsigned short*)rightT + (size_t)(d0 + nn) * NNROWS + row0 + 8 * half) = o;
    }
    // gate = sigmoid(p@ogw.T + ogb), natural [row][d] layout
    {
        v8f aO = gemm_row128(a, Wog + (size_t)d0 * DDIM, lane);
        float bo = ogb[d0 + nn];
        unsigned short* g = (unsigned short*)gate;
#pragma unroll
        for (int r = 0; r < 8; ++r) {
            int m = 8 * half + r;
            g[(row0 + m) * DDIM + d0 + nn] = f2bf(sigf(aO[r] + bo));
        }
    }
}

// ---------------- K3: triangular per-channel matmul out[i,j,d] = sum_k L[d,i,k]*R[d,j,k]
// grid (N/32, N/32, D/8); each wave: one channel, 32x32 (i x j) register tile
// (2x2 WMMA tiles, A/B fragments reused -> 2 b128 loads per WMMA instead of 4).
// LDS re-layout for contiguous [i,j,d] stores.
__global__ void k_tri_matmul(const bf16_t* __restrict__ leftT,
                             const bf16_t* __restrict__ rightT,
                             float* __restrict__ out) {
    __shared__ float tile[32 * 32 * 8];   // 32 KB
    int wv   = threadIdx.x >> 5;
    int lane = threadIdx.x & 31;
    int half = lane >> 4;
    int nn   = lane & 15;
    int i0 = blockIdx.x * 32;
    int j0 = blockIdx.y * 32;
    int dg = blockIdx.z * 8;
    int d  = dg + wv;

    const bf16_t* L = leftT  + (size_t)d * NNROWS + (size_t)i0 * NDIM;
    const bf16_t* R = rightT + (size_t)d * NNROWS + (size_t)j0 * NDIM;

    v8f acc00 = {0.f, 0.f, 0.f, 0.f, 0.f, 0.f, 0.f, 0.f};
    v8f acc01 = {0.f, 0.f, 0.f, 0.f, 0.f, 0.f, 0.f, 0.f};
    v8f acc10 = {0.f, 0.f, 0.f, 0.f, 0.f, 0.f, 0.f, 0.f};
    v8f acc11 = {0.f, 0.f, 0.f, 0.f, 0.f, 0.f, 0.f, 0.f};

    for (int k0 = 0; k0 < NDIM; k0 += 32) {
        __builtin_prefetch((const void*)(L + k0 + 128), 0, 0);
        __builtin_prefetch((const void*)(R + k0 + 128), 0, 0);
        v16bf a0 = load_a_frag(L + k0,             NDIM, lane);
        v16bf a1 = load_a_frag(L + 16 * NDIM + k0, NDIM, lane);
        v16bf b0 = load_b_frag(R + k0,             NDIM, lane);
        v16bf b1 = load_b_frag(R + 16 * NDIM + k0, NDIM, lane);
        acc00 = wmma_bf16(a0, b0, acc00);
        acc01 = wmma_bf16(a0, b1, acc01);
        acc10 = wmma_bf16(a1, b0, acc10);
        acc11 = wmma_bf16(a1, b1, acc11);
    }

#pragma unroll
    for (int r = 0; r < 8; ++r) {
        int m = 8 * half + r;
        tile[((m)      * 32 + nn)      * 8 + wv] = acc00[r];
        tile[((m)      * 32 + nn + 16) * 8 + wv] = acc01[r];
        tile[((m + 16) * 32 + nn)      * 8 + wv] = acc10[r];
        tile[((m + 16) * 32 + nn + 16) * 8 + wv] = acc11[r];
    }
    __syncthreads();

    // 1024 (m,n) pairs; 4 per thread; each writes 8 floats (32B contiguous in d)
#pragma unroll
    for (int t = 0; t < 4; ++t) {
        int pp = threadIdx.x + t * 256;
        int m = pp >> 5;
        int n = pp & 31;
        float* dst = out + ((size_t)(i0 + m) * NDIM + (j0 + n)) * DDIM + dg;
        const float4* src = (const float4*)&tile[(m * 32 + n) * 8];
        ((float4*)dst)[0] = src[0];
        ((float4*)dst)[1] = src[1];
    }
}

// ---------------- K4b: out = (LN(omm)@opw.T + opb) * gate ----------------
__global__ void k_out_proj(const bf16_t* __restrict__ q,
                           const bf16_t* __restrict__ Wop,
                           const float* __restrict__ opb,
                           const bf16_t* __restrict__ gate,
                           float* __restrict__ out) {
    int wv   = threadIdx.x >> 5;
    int lane = threadIdx.x & 31;
    int half = lane >> 4;
    int nn   = lane & 15;
    int d0   = wv * 16;
    size_t row0 = (size_t)blockIdx.x * 16;

    v16bf a[4];
#pragma unroll
    for (int ks = 0; ks < 4; ++ks)
        a[ks] = load_a_frag(q + row0 * DDIM + ks * 32, DDIM, lane);

    v8f acc = gemm_row128(a, Wop + (size_t)d0 * DDIM, lane);
    float bo = opb[d0 + nn];
    const unsigned short* g = (const unsigned short*)gate;
#pragma unroll
    for (int r = 0; r < 8; ++r) {
        int m = 8 * half + r;
        size_t idx = (row0 + m) * DDIM + d0 + nn;
        out[idx] = (acc[r] + bo) * bf2f(g[idx]);
    }
}

// ---------------- host launcher ----------------
extern "C" void kernel_launch(void* const* d_in, const int* in_sizes, int n_in,
                              void* d_out, int out_size, void* d_ws, size_t ws_size,
                              hipStream_t stream) {
    (void)in_sizes; (void)n_in; (void)out_size; (void)ws_size;
    const float* pair   = (const float*)d_in[0];
    const float* mask   = (const float*)d_in[1];
    const float* norm_w = (const float*)d_in[2];
    const float* norm_b = (const float*)d_in[3];
    const float* lpw = (const float*)d_in[4];   const float* lpb = (const float*)d_in[5];
    const float* lgw = (const float*)d_in[6];   const float* lgb = (const float*)d_in[7];
    const float* rpw = (const float*)d_in[8];   const float* rpb = (const float*)d_in[9];
    const float* rgw = (const float*)d_in[10];  const float* rgb = (const float*)d_in[11];
    const float* onw = (const float*)d_in[12];  const float* onb = (const float*)d_in[13];
    const float* opw = (const float*)d_in[14];  const float* opb = (const float*)d_in[15];
    const float* ogw = (const float*)d_in[16];  const float* ogb = (const float*)d_in[17];

    char* ws = (char*)d_ws;
    const size_t MB = 1024ull * 1024ull;
    bf16_t* p_bf   = (bf16_t*)(ws);              // 64 MiB (reused as q after K3)
    bf16_t* leftT  = (bf16_t*)(ws + 64 * MB);    // 64 MiB, layout [d][row]
    bf16_t* rightT = (bf16_t*)(ws + 128 * MB);   // 64 MiB, layout [d][row]
    bf16_t* gate   = (bf16_t*)(ws + 192 * MB);   // 64 MiB, layout [row][d]
    float*  omm    = (float*)(ws + 256 * MB);    // 128 MiB, layout [i][j][d]
    bf16_t* wbf    = (bf16_t*)(ws + 384 * MB);   // 6 x 32 KiB bf16 weights

    // K0: weight conversion (6 * 16384 elems)
    k_cvt_weights<<<384, 256, 0, stream>>>(lpw, lgw, rpw, rgw, opw, ogw, wbf);
    // K1: p = LN(pair) -> bf16
    k_layernorm_bf16<<<NNROWS / 8, 256, 0, stream>>>(pair, norm_w, norm_b, p_bf);
    // K2: gated projections -> leftT, rightT (transposed), gate
    k_proj_gated<<<NNROWS / 16, 256, 0, stream>>>(p_bf, wbf, lpb, lgb, rpb, rgb, ogb,
                                                  mask, leftT, rightT, gate);
    // K3: triangular per-channel matmul -> omm [i,j,d]
    k_tri_matmul<<<dim3(16, 16, 16), 256, 0, stream>>>(leftT, rightT, omm);
    // K4a: q = LN(omm) -> bf16 (reuses p_bf slot)
    k_layernorm_bf16<<<NNROWS / 8, 256, 0, stream>>>(omm, onw, onb, p_bf);
    // K4b: out = (q@opw.T + opb) * gate
    k_out_proj<<<NNROWS / 16, 256, 0, stream>>>(p_bf, wbf + (size_t)4 * 16384, opb,
                                                gate, (float*)d_out);
}
